// MultiHeadDotProductAttention_80676665688651
// MI455X (gfx1250) — compile-verified, matrix operand order checked by reading, not run on hardware
//
#include <hip/hip_runtime.h>
#include <hip/hip_bf16.h>
#include <cstdint>
#include <cstddef>

#define DEV __device__ __forceinline__

typedef __attribute__((ext_vector_type(16))) __bf16        v16bf;
typedef __attribute__((ext_vector_type(8)))  float         v8f;
typedef __attribute__((ext_vector_type(4)))  unsigned int  v4u;
typedef __attribute__((ext_vector_type(2)))  unsigned int  v2u;
typedef __attribute__((ext_vector_type(4)))  float         v4f;

union FragBF {
  v16bf v;
  v4u   q[2];
};

DEV unsigned short f32_to_bf16(float f) {
  unsigned int u = __float_as_uint(f);
  unsigned int r = u + 0x7FFFu + ((u >> 16) & 1u);   // round-to-nearest-even
  return (unsigned short)(r >> 16);
}

DEV v4u pack8_bf16(const float* v) {
  v4u o;
  o.x = (unsigned)f32_to_bf16(v[0]) | ((unsigned)f32_to_bf16(v[1]) << 16);
  o.y = (unsigned)f32_to_bf16(v[2]) | ((unsigned)f32_to_bf16(v[3]) << 16);
  o.z = (unsigned)f32_to_bf16(v[4]) | ((unsigned)f32_to_bf16(v[5]) << 16);
  o.w = (unsigned)f32_to_bf16(v[6]) | ((unsigned)f32_to_bf16(v[7]) << 16);
  return o;
}

DEV v8f zero8() { v8f z = {0.f,0.f,0.f,0.f,0.f,0.f,0.f,0.f}; return z; }

// A / B^T fragment: lane (l<16) holds row M=l, K in {k0..k0+7, k0+16..k0+23};
// lane (l>=16) same row set, K shifted by 8. Two 16B loads per fragment.
DEV v16bf load_frag_g(const unsigned short* __restrict__ base, int ld, int row0, int k0) {
  const int lane = threadIdx.x & 31;
  const size_t r = (size_t)(row0 + (lane & 15));
  const int   kb = k0 + ((lane >> 4) << 3);
  FragBF f;
  const unsigned short* p = base + r * (size_t)ld + kb;
  f.q[0] = *(const v4u*)(p);
  f.q[1] = *(const v4u*)(p + 16);
  return f.v;
}

// Fragment from an LDS tile stored row-major [128][32] bf16
DEV v16bf load_frag_lds(const unsigned short* base, int row0) {
  const int lane = threadIdx.x & 31;
  const unsigned short* p = base + (size_t)(row0 + (lane & 15)) * 32 + ((lane >> 4) << 3);
  FragBF f;
  f.q[0] = *(const v4u*)(p);
  f.q[1] = *(const v4u*)(p + 16);
  return f.v;
}

DEV v8f wmma_bf16(v16bf a, v16bf b, v8f c) {
  return __builtin_amdgcn_wmma_f32_16x16x32_bf16(false, a, false, b, (short)0, c, false, false);
}

// Async copy of one 16B chunk global -> LDS (ASYNCcnt-tracked, CDNA5)
DEV void async_b128(unsigned lds_off, const unsigned short* g) {
  asm volatile("global_load_async_to_lds_b128 %0, %1, off"
               :: "v"(lds_off), "v"((unsigned long long)(uintptr_t)g)
               : "memory");
}
DEV void wait_async0() { asm volatile("s_wait_asynccnt 0" ::: "memory"); }

// ---------------------------------------------------------------------------
// 1) x (f32) -> bf16, vectorized x4
// ---------------------------------------------------------------------------
__global__ __launch_bounds__(256) void conv_f32_bf16_vec(
    const float* __restrict__ in, unsigned short* __restrict__ out, int n4) {
  int i = blockIdx.x * 256 + threadIdx.x;
  if (i >= n4) return;
  v4f f = ((const v4f*)in)[i];
  v2u o;
  o.x = (unsigned)f32_to_bf16(f.x) | ((unsigned)f32_to_bf16(f.y) << 16);
  o.y = (unsigned)f32_to_bf16(f.z) | ((unsigned)f32_to_bf16(f.w) << 16);
  ((v2u*)out)[i] = o;
}

// ---------------------------------------------------------------------------
// 2) 2048x2048 f32 transpose -> bf16 (LDS-tiled), 4 matrices via blockIdx.z
// ---------------------------------------------------------------------------
__global__ __launch_bounds__(256) void transpose_bf16_2048(
    const float* __restrict__ in0, const float* __restrict__ in1,
    const float* __restrict__ in2, const float* __restrict__ in3,
    unsigned short* __restrict__ o0, unsigned short* __restrict__ o1,
    unsigned short* __restrict__ o2, unsigned short* __restrict__ o3) {
  __shared__ float tile[32][33];
  const float* in = (blockIdx.z == 0) ? in0 : (blockIdx.z == 1) ? in1
                   : (blockIdx.z == 2) ? in2 : in3;
  unsigned short* out = (blockIdx.z == 0) ? o0 : (blockIdx.z == 1) ? o1
                       : (blockIdx.z == 2) ? o2 : o3;
  const int tx = threadIdx.x, ty = threadIdx.y;       // 32 x 8
  const int x  = blockIdx.x * 32 + tx;
  const int y0 = blockIdx.y * 32;
  #pragma unroll
  for (int i = 0; i < 32; i += 8)
    tile[ty + i][tx] = in[(size_t)(y0 + ty + i) * 2048 + x];
  __syncthreads();
  const int ox  = y0 + tx;
  const int oy0 = blockIdx.x * 32;
  #pragma unroll
  for (int i = 0; i < 32; i += 8)
    out[(size_t)(oy0 + ty + i) * 2048 + ox] = f32_to_bf16(tile[tx][ty + i]);
}

// ---------------------------------------------------------------------------
// 3) Unified LDS-staged GEMM (bf16 WMMA, f32 acc), async double-buffered.
//    Block computes a 128x128 tile; 8 waves: wave w -> rows (w>>1)*32..+31,
//    cols (w&1)*64..+63 (8 WMMA tiles). K loop: 64 chunks of 32; A/B 128x32
//    tiles staged with global_load_async_to_lds_b128 (ASYNCcnt), double
//    buffered. k-loop is unroll(1) so accumulators stay pinned in place
//    (avoids cross-parity v_mov shuffles + hazard NOPs).
//    mode 0: Q = x@wq, fused RoPE + 1/sqrt(H) -> outb [B][N][S][H]
//    mode 1: K = x@wk, fused RoPE             -> outb [B][N][S][H]
//    mode 2: V = x@wv                         -> outb [B][N][H][S] (transposed)
//    mode 3: out = attn@wo                    -> outf [B*S][D] f32
// ---------------------------------------------------------------------------
__global__ __launch_bounds__(256) void gemm_bf16_lds(
    const unsigned short* __restrict__ Amat,  // [8192][2048] bf16
    const unsigned short* __restrict__ Bmat,  // [2048][2048] bf16 (B transposed)
    unsigned short* __restrict__ outb,
    float* __restrict__ outf, int mode) {
  __shared__ __align__(16) unsigned short Atile[2][128 * 32];
  __shared__ __align__(16) unsigned short Btile[2][128 * 32];

  const int t    = threadIdx.x;
  const int lane = t & 31;
  const int half = lane >> 4;
  const int w    = t >> 5;
  const int rgroup = w >> 1;            // 0..3 -> row offset rgroup*32
  const int cgroup = w & 1;             // 0..1 -> col offset cgroup*64

  const int rowblock = blockIdx.x >> 4; // 64 row blocks
  const int colblock = blockIdx.x & 15; // 16 col blocks
  const int row0 = rowblock * 128;
  const int col0 = colblock * 128;

  const unsigned a0off = (unsigned)(uintptr_t)&Atile[0][0];
  const unsigned a1off = (unsigned)(uintptr_t)&Atile[1][0];
  const unsigned b0off = (unsigned)(uintptr_t)&Btile[0][0];
  const unsigned b1off = (unsigned)(uintptr_t)&Btile[1][0];

  const unsigned short* gA = Amat + (size_t)row0 * 2048;
  const unsigned short* gB = Bmat + (size_t)col0 * 2048;

  // stage one 128x32 A tile + 128x32 B tile (8KB each) into buffer `buf`
  auto stage = [&](int k0, int buf) {
    const unsigned abase = buf ? a1off : a0off;
    const unsigned bbase = buf ? b1off : b0off;
    #pragma unroll
    for (int i = 0; i < 2; ++i) {
      const int chunk = t + 256 * i;          // 512 chunks of 16B
      const int row = chunk >> 2;
      const int co  = (chunk & 3) << 3;       // element offset within row
      async_b128(abase + (unsigned)chunk * 16u, gA + (size_t)row * 2048 + k0 + co);
      async_b128(bbase + (unsigned)chunk * 16u, gB + (size_t)row * 2048 + k0 + co);
    }
  };

  v8f acc[2][4];
  #pragma unroll
  for (int rt = 0; rt < 2; ++rt)
    #pragma unroll
    for (int j = 0; j < 4; ++j) acc[rt][j] = zero8();

  stage(0, 0);
  wait_async0();
  __syncthreads();

  #pragma unroll 1
  for (int kk = 0; kk < 64; ++kk) {
    const int cur = kk & 1;
    if (kk + 1 < 64) stage((kk + 1) * 32, cur ^ 1);

    const unsigned short* At = &Atile[cur][0];
    const unsigned short* Bt = &Btile[cur][0];
    v16bf a0 = load_frag_lds(At, rgroup * 32);
    v16bf a1 = load_frag_lds(At, rgroup * 32 + 16);
    #pragma unroll
    for (int j = 0; j < 4; ++j) {
      v16bf b = load_frag_lds(Bt, cgroup * 64 + j * 16);
      acc[0][j] = wmma_bf16(a0, b, acc[0][j]);
      acc[1][j] = wmma_bf16(a1, b, acc[1][j]);
    }
    wait_async0();       // own prefetch done; barrier makes all waves' data visible
    __syncthreads();
  }

  // ---------------- epilogue ----------------
  const int b = row0 >> 11;                    // S = 2048
  #pragma unroll
  for (int rt = 0; rt < 2; ++rt) {
    const int gr0 = row0 + rgroup * 32 + rt * 16;  // tile's first global row
    const int s0  = gr0 & 2047;
    #pragma unroll
    for (int j = 0; j < 4; ++j) {
      const int col = col0 + cgroup * 64 + j * 16 + (lane & 15);
      const int n   = col >> 7;
      const int h   = col & 127;
      float vals[8];
      #pragma unroll
      for (int r = 0; r < 8; ++r) vals[r] = acc[rt][j][r];

      if (mode <= 1) {  // fused LLaMA RoPE (interleaved pairs = adjacent lanes)
        const float inv_ts =
            __expf(-9.210340371976184f * (float)((h >> 1) * 2) * (1.0f / 128.0f));
        #pragma unroll
        for (int r = 0; r < 8; ++r) {
          float xv = vals[r];
          float pv = __shfl_xor(xv, 1, 32);
          float s  = (float)(s0 + r + 8 * half);
          float sn, cs;
          __sincosf(s * inv_ts, &sn, &cs);
          float o = (lane & 1) ? (xv * cs + pv * sn) : (xv * cs - pv * sn);
          if (mode == 0) o *= 0.08838834764831845f;   // 1/sqrt(128)
          vals[r] = o;
        }
      }

      if (mode == 2) {
        // V^T: [B][N][H][S]; 8 consecutive s per lane -> one b128 store
        unsigned short* dst =
            outb + (((size_t)(b * 16 + n) * 128 + h) * 2048 + s0 + 8 * half);
        *(v4u*)dst = pack8_bf16(vals);
      } else if (mode == 3) {
        #pragma unroll
        for (int r = 0; r < 8; ++r)
          outf[(size_t)(gr0 + r + 8 * half) * 2048 + col] = vals[r];
      } else {
        // [B][N][S][H], stride H between rows
        unsigned short* dst =
            outb + (((size_t)(b * 16 + n) * 2048 + (s0 + 8 * half)) * 128 + h);
        #pragma unroll
        for (int r = 0; r < 8; ++r) dst[(size_t)r * 128] = f32_to_bf16(vals[r]);
      }
    }
  }
}

// ---------------------------------------------------------------------------
// 4) Causal flash attention: one wave per 16-query tile of one (b,n).
//    Per 32-key block: 8 WMMA (QK^T), online softmax in registers,
//    P -> LDS -> A-fragment (s_wait_dscnt), 8 WMMA (P*V) into 16x128 f32 acc.
// ---------------------------------------------------------------------------
__global__ __launch_bounds__(256) void flash_attn(
    const unsigned short* __restrict__ Qb,   // [B*N][S][H]
    const unsigned short* __restrict__ Kb,   // [B*N][S][H]
    const unsigned short* __restrict__ Vtb,  // [B*N][H][S]
    unsigned short* __restrict__ Ob) {       // [B*S][NH]
  __shared__ __align__(16) unsigned short plds[8][16 * 32];
  const int lane = threadIdx.x & 31;
  const int half = lane >> 4;
  const int widx = threadIdx.x >> 5;
  const int wave = blockIdx.x * 8 + widx;
  const int qtile = wave & 127;              // S/16
  const int bn    = wave >> 7;
  const int q0    = qtile * 16;

  const unsigned short* Q  = Qb  + (size_t)bn * 2048 * 128;
  const unsigned short* K  = Kb  + (size_t)bn * 2048 * 128;
  const unsigned short* Vt = Vtb + (size_t)bn * 128 * 2048;

  v16bf qf[4];
  #pragma unroll
  for (int c = 0; c < 4; ++c) qf[c] = load_frag_g(Q, 128, q0, c * 32);

  v8f o[8];
  #pragma unroll
  for (int tt = 0; tt < 8; ++tt) o[tt] = zero8();
  float rm[8], rs[8];
  #pragma unroll
  for (int r = 0; r < 8; ++r) { rm[r] = -__builtin_inff(); rs[r] = 0.f; }

  unsigned short* P = &plds[widx][0];

  #pragma unroll 1
  for (int kb = 0; kb <= q0; kb += 32) {     // covers keys 0..q0+15 (causal)
    v8f sacc0 = zero8(), sacc1 = zero8();
    #pragma unroll
    for (int c = 0; c < 4; ++c) {
      v16bf b0 = load_frag_g(K, 128, kb,      c * 32);
      v16bf b1 = load_frag_g(K, 128, kb + 16, c * 32);
      sacc0 = wmma_bf16(qf[c], b0, sacc0);
      sacc1 = wmma_bf16(qf[c], b1, sacc1);
    }
    const int k0i = kb + (lane & 15);
    #pragma unroll
    for (int r = 0; r < 8; ++r) {
      const int q = q0 + r + 8 * half;
      float v0 = sacc0[r];
      float v1 = sacc1[r];
      if (k0i > q)      v0 = -1e30f;
      if (k0i + 16 > q) v1 = -1e30f;
      float bm = fmaxf(v0, v1);
      #pragma unroll
      for (int m = 1; m <= 8; m <<= 1) bm = fmaxf(bm, __shfl_xor(bm, m, 32));
      const float mnew  = fmaxf(rm[r], bm);
      const float alpha = __expf(rm[r] - mnew);
      const float p0 = __expf(v0 - mnew);
      const float p1 = __expf(v1 - mnew);
      float ls = p0 + p1;
      #pragma unroll
      for (int m = 1; m <= 8; m <<= 1) ls += __shfl_xor(ls, m, 32);
      rs[r] = rs[r] * alpha + ls;
      rm[r] = mnew;
      #pragma unroll
      for (int tt = 0; tt < 8; ++tt) o[tt][r] *= alpha;
      P[(r + 8 * half) * 32 + (lane & 15)]      = f32_to_bf16(p0);
      P[(r + 8 * half) * 32 + 16 + (lane & 15)] = f32_to_bf16(p1);
    }
    asm volatile("s_wait_dscnt 0" ::: "memory");   // P stores visible to re-layout loads
    FragBF pf;
    {
      const unsigned short* pp = P + (size_t)(lane & 15) * 32 + ((lane >> 4) << 3);
      pf.q[0] = *(const v4u*)(pp);
      pf.q[1] = *(const v4u*)(pp + 16);
    }
    #pragma unroll
    for (int tt = 0; tt < 8; ++tt) {
      v16bf vb = load_frag_g(Vt, 2048, tt * 16, kb);
      o[tt] = wmma_bf16(pf.v, vb, o[tt]);
    }
  }

  const int b = bn >> 4, n = bn & 15;
  #pragma unroll
  for (int r = 0; r < 8; ++r) {
    const float inv = 1.0f / rs[r];
    const size_t s  = (size_t)q0 + r + 8 * half;
    unsigned short* dst =
        Ob + ((size_t)b * 2048 + s) * 2048 + n * 128 + (lane & 15);
    #pragma unroll
    for (int tt = 0; tt < 8; ++tt) dst[tt * 16] = f32_to_bf16(o[tt][r] * inv);
  }
}

// ---------------------------------------------------------------------------
// Host launcher
// ---------------------------------------------------------------------------
extern "C" void kernel_launch(void* const* d_in, const int* in_sizes, int n_in,
                              void* d_out, int out_size, void* d_ws, size_t ws_size,
                              hipStream_t stream) {
  (void)in_sizes; (void)n_in; (void)out_size; (void)ws_size;
  const float* x  = (const float*)d_in[0];
  const float* wq = (const float*)d_in[1];
  const float* wk = (const float*)d_in[2];
  const float* wv = (const float*)d_in[3];
  const float* wo = (const float*)d_in[4];
  float* out = (float*)d_out;

  // B=4, S=2048, D=2048, N=16, H=128, NH=2048, B*S=8192
  char* ws = (char*)d_ws;
  size_t off = 0;
  auto alloc = [&](size_t bytes) {
    char* p = ws + off;
    off += (bytes + 255) & ~(size_t)255;
    return p;
  };
  const size_t BSD2 = (size_t)8192 * 2048 * 2;   // 33.5 MB (bf16 [B*S][2048])
  const size_t W2   = (size_t)2048 * 2048 * 2;   //  8.4 MB

  unsigned short* xb  = (unsigned short*)alloc(BSD2);
  unsigned short* wqt = (unsigned short*)alloc(W2);
  unsigned short* wkt = (unsigned short*)alloc(W2);
  unsigned short* wvt = (unsigned short*)alloc(W2);
  unsigned short* wot = (unsigned short*)alloc(W2);
  unsigned short* Qb  = (unsigned short*)alloc(BSD2);
  unsigned short* Kb  = (unsigned short*)alloc(BSD2);
  unsigned short* Vtb = (unsigned short*)alloc(BSD2);
  unsigned short* Ab  = (unsigned short*)alloc(BSD2);

  // Stage to bf16 (+ weight transposes for contiguous B^T fragments)
  conv_f32_bf16_vec<<<4194304 / 256, 256, 0, stream>>>(x, xb, 4194304);
  transpose_bf16_2048<<<dim3(64, 64, 4), dim3(32, 8), 0, stream>>>(
      wq, wk, wv, wo, wqt, wkt, wvt, wot);

  // QKV projections + fused RoPE / scaling / attention layouts (async LDS GEMM)
  gemm_bf16_lds<<<1024, 256, 0, stream>>>(xb, wqt, Qb, nullptr, 0);
  gemm_bf16_lds<<<1024, 256, 0, stream>>>(xb, wkt, Kb, nullptr, 1);
  gemm_bf16_lds<<<1024, 256, 0, stream>>>(xb, wvt, Vtb, nullptr, 2);

  // Causal flash attention: 8192 waves (B*N*S/16), 8 waves/block
  flash_attn<<<1024, 256, 0, stream>>>(Qb, Kb, Vtb, Ab);

  // Output projection to f32 (async LDS GEMM)
  gemm_bf16_lds<<<1024, 256, 0, stream>>>(Ab, wot, nullptr, out, 3);
}
